// ManyToOne_RNN_34943853920808
// MI455X (gfx1250) — matrix operand order repaired and checked by reading, not compiled, not run on hardware
//
#include <hip/hip_runtime.h>
#include <hip/hip_bf16.h>

// ---------------------------------------------------------------------------
// Problem constants (from the reference)
// ---------------------------------------------------------------------------
#define ISZ   512          // INPUT_SIZE
#define HSZ   1024         // HIDDEN_SIZE
#define LSTEPS 256         // L
#define NB    512          // N (batch)
#define KCAT  (ISZ + HSZ)  // 1536 concatenated K for fused step GEMM
#define KTILE 64           // K staged per LDS buffer (2 WMMA chunks)

typedef __attribute__((ext_vector_type(16))) __bf16 v16bf;
typedef __attribute__((ext_vector_type(8)))  float  v8f;

// ---------------------------------------------------------------------------
// Small helper kernels (pack / convert)
// ---------------------------------------------------------------------------

// Wcat[n][k] (bf16, ld=KCAT): k<ISZ -> W_ih[n][k], else W_hh[n][k-ISZ]
__global__ __launch_bounds__(256) void pack_wcat_kernel(
    const float* __restrict__ W_ih, const float* __restrict__ W_hh,
    __bf16* __restrict__ Wcat)
{
    int idx = blockIdx.x * blockDim.x + threadIdx.x;
    int total = HSZ * KCAT;
    if (idx >= total) return;
    int n = idx / KCAT;
    int k = idx % KCAT;
    float v = (k < ISZ) ? W_ih[(size_t)n * ISZ + k]
                        : W_hh[(size_t)n * HSZ + (k - ISZ)];
    Wcat[idx] = (__bf16)v;
}

__global__ __launch_bounds__(256) void pack_bias_kernel(
    const float* __restrict__ b_ih, const float* __restrict__ b_hh,
    float* __restrict__ bsum)
{
    int idx = blockIdx.x * blockDim.x + threadIdx.x;
    if (idx < HSZ) bsum[idx] = b_ih[idx] + b_hh[idx];
}

// Generic strided fp32 -> bf16 tile convert (used for x_t head, h0 tail, W_fc)
__global__ __launch_bounds__(256) void cvt_f32_bf16_kernel(
    const float* __restrict__ src, int sld,
    __bf16* __restrict__ dst, int dld,
    int rows, int cols)
{
    int idx = blockIdx.x * blockDim.x + threadIdx.x;
    int total = rows * cols;
    if (idx >= total) return;
    int r = idx / cols;
    int c = idx % cols;
    dst[(size_t)r * dld + c] = (__bf16)src[(size_t)r * sld + c];
}

// ---------------------------------------------------------------------------
// WMMA GEMM + activation:
//   C[m][n] = act( sum_k A[m][k] * W[n][k] + bias[n] )
// A: bf16 [M x K] row-major (lda), W: bf16 [Nout x K] row-major (ldw).
// Block: 128 threads (4 waves), 64x64 output tile. Wave: 32x32 (2x2 WMMA).
// K staged in KTILE=64 slabs with double-buffered LDS: ONE barrier per slab,
// global loads + ds stores for slab s+1 overlap the 8 WMMAs of slab s.
// act: 0 = tanh, 1 = relu.
// M, Nout multiples of 64; K multiple of 64.
// ---------------------------------------------------------------------------
__global__ __launch_bounds__(128) void gemm_act_kernel(
    const __bf16* __restrict__ A, int lda,
    const __bf16* __restrict__ W, int ldw,
    const float* __restrict__ bias,
    int K, int act,
    __bf16* __restrict__ out_bf, int ldob,
    float* __restrict__ out_f, int ldof)
{
    __shared__ __bf16 As[2][64][KTILE];   // 16 KB : rows (M) x K slab, x2 bufs
    __shared__ __bf16 Ws[2][64][KTILE];   // 16 KB : cols (N) x K slab, x2 bufs

    const int tid  = threadIdx.x;
    const int lane = tid & 31;
    const int wave = tid >> 5;
    const int wm   = wave >> 1;      // 0..1  (M sub-tile of the block tile)
    const int wn   = wave & 1;       // 0..1  (N sub-tile)
    const int kh   = lane >> 4;      // lane half (0/1)
    const int l16  = lane & 15;

    const int rowBase = blockIdx.y * 64;
    const int colBase = blockIdx.x * 64;

    // --- accumulators, initialized with bias broadcast ---------------------
    // C/D layout (ISA 7.12.2): element (VGPR r, lane) = (M = kh*8 + r, N = l16)
    v8f acc[2][2];
    for (int nt = 0; nt < 2; ++nt) {
        float b = bias ? bias[colBase + wn * 32 + nt * 16 + l16] : 0.0f;
        for (int mt = 0; mt < 2; ++mt)
            for (int r = 0; r < 8; ++r)
                acc[mt][nt][r] = b;
    }

    // --- cooperative slab load: 64 rows x 64 K = 512 uint4 per array -------
    // 128 threads x 4 uint4 (16B) each, fully coalesced.
    uint4 ra[4], rw[4];
    auto loadStage = [&](int k0) {
        #pragma unroll
        for (int q = 0; q < 4; ++q) {
            int idx = q * 128 + tid;
            int row = idx >> 3;           // 0..63
            int col = (idx & 7) * 8;      // 0..56 step 8 (bf16 elems)
            ra[q] = *(const uint4*)(A + (size_t)(rowBase + row) * lda + k0 + col);
            rw[q] = *(const uint4*)(W + (size_t)(colBase + row) * ldw + k0 + col);
        }
    };
    auto storeStage = [&](int buf) {
        #pragma unroll
        for (int q = 0; q < 4; ++q) {
            int idx = q * 128 + tid;
            int row = idx >> 3;
            int col = (idx & 7) * 8;
            *(uint4*)&As[buf][row][col] = ra[q];
            *(uint4*)&Ws[buf][row][col] = rw[q];
        }
    };

    union Frag { uint4 q[2]; v16bf v; };

    const int nStages = K / KTILE;
    loadStage(0);
    storeStage(0);

    for (int s = 0; s < nStages; ++s) {
        const int buf = s & 1;
        __syncthreads();   // slab `buf` visible; prior reads of buf^1 done
        if (s + 1 < nStages) {
            loadStage((s + 1) * KTILE);
            storeStage(buf ^ 1);          // fill other buffer while computing
        }

        #pragma unroll
        for (int c = 0; c < 2; ++c) {     // two 32-deep WMMA chunks per slab
            const int kb = c * 32;
            // A fragments (16x32, ISA layout): elems 0..7 -> K = kb+kh*8+j,
            //                                  elems 8..15 -> K = kb+16+kh*8+j
            v16bf afrag[2];
            #pragma unroll
            for (int mt = 0; mt < 2; ++mt) {
                int m = wm * 32 + mt * 16 + l16;
                Frag f;
                f.q[0] = *(const uint4*)&As[buf][m][kb + kh * 8];
                f.q[1] = *(const uint4*)&As[buf][m][kb + 16 + kh * 8];
                afrag[mt] = f.v;
            }
            // B fragments (32x16): elems 0..7 -> K = kb+kh*8+j, elems 8..15 -> K = kb+16+kh*8+j, col=l16
            v16bf wfrag[2];
            #pragma unroll
            for (int nt = 0; nt < 2; ++nt) {
                int n = wn * 32 + nt * 16 + l16;
                Frag f;
                f.q[0] = *(const uint4*)&Ws[buf][n][kb + kh * 8];
                f.q[1] = *(const uint4*)&Ws[buf][n][kb + 16 + kh * 8];
                wfrag[nt] = f.v;
            }

            #pragma unroll
            for (int mt = 0; mt < 2; ++mt)
                #pragma unroll
                for (int nt = 0; nt < 2; ++nt)
                    acc[mt][nt] = __builtin_amdgcn_wmma_f32_16x16x32_bf16(
                        /*neg_a=*/false, afrag[mt],
                        /*neg_b=*/false, wfrag[nt],
                        /*c_mod=*/(short)0, acc[mt][nt],
                        /*reuse_a=*/false, /*reuse_b=*/false);
        }
    }

    // --- epilogue: activation + stores -------------------------------------
    #pragma unroll
    for (int mt = 0; mt < 2; ++mt) {
        #pragma unroll
        for (int nt = 0; nt < 2; ++nt) {
            int n = colBase + wn * 32 + nt * 16 + l16;
            #pragma unroll
            for (int r = 0; r < 8; ++r) {
                int m = rowBase + wm * 32 + mt * 16 + kh * 8 + r;
                float v = acc[mt][nt][r];
                v = (act == 0) ? tanhf(v) : fmaxf(v, 0.0f);
                if (out_bf) out_bf[(size_t)m * ldob + n] = (__bf16)v;
                if (out_f)  out_f [(size_t)m * ldof + n] = v;
            }
        }
    }
}

// ---------------------------------------------------------------------------
// Host-side orchestration
// ---------------------------------------------------------------------------
extern "C" void kernel_launch(void* const* d_in, const int* in_sizes, int n_in,
                              void* d_out, int out_size, void* d_ws, size_t ws_size,
                              hipStream_t stream)
{
    const float* x    = (const float*)d_in[0];  // [L, N, ISZ]
    const float* h0   = (const float*)d_in[1];  // [N, HSZ]
    const float* W_ih = (const float*)d_in[2];  // [HSZ, ISZ]
    const float* W_hh = (const float*)d_in[3];  // [HSZ, HSZ]
    const float* b_ih = (const float*)d_in[4];  // [HSZ]
    const float* b_hh = (const float*)d_in[5];  // [HSZ]
    const float* W_fc = (const float*)d_in[6];  // [ISZ, HSZ]
    const float* b_fc = (const float*)d_in[7];  // [ISZ]

    float* y_out = (float*)d_out;                      // [N, ISZ]
    float* h_out = (float*)d_out + (size_t)NB * ISZ;   // [N, HSZ]

    // workspace layout (all 16B-aligned sizes), ~7.3 MB total
    char* ws = (char*)d_ws;
    __bf16* Wcat  = (__bf16*)ws;  ws += (size_t)HSZ * KCAT * sizeof(__bf16); // 3 MB
    __bf16* Wfc_b = (__bf16*)ws;  ws += (size_t)ISZ * HSZ  * sizeof(__bf16); // 1 MB
    float*  bsum  = (float*) ws;  ws += (size_t)HSZ * sizeof(float);         // 4 KB
    __bf16* AcatA = (__bf16*)ws;  ws += (size_t)NB * KCAT * sizeof(__bf16);  // 1.5 MB
    __bf16* AcatB = (__bf16*)ws;  ws += (size_t)NB * KCAT * sizeof(__bf16);  // 1.5 MB

    // --- one-time packs ----------------------------------------------------
    {
        int total = HSZ * KCAT;
        pack_wcat_kernel<<<(total + 255) / 256, 256, 0, stream>>>(W_ih, W_hh, Wcat);
    }
    {
        int total = ISZ * HSZ;
        cvt_f32_bf16_kernel<<<(total + 255) / 256, 256, 0, stream>>>(
            W_fc, HSZ, Wfc_b, HSZ, ISZ, HSZ);
    }
    pack_bias_kernel<<<(HSZ + 255) / 256, 256, 0, stream>>>(b_ih, b_hh, bsum);
    {
        // h0 (fp32) -> bf16 tail of AcatA (cols ISZ..KCAT)
        int total = NB * HSZ;
        cvt_f32_bf16_kernel<<<(total + 255) / 256, 256, 0, stream>>>(
            h0, HSZ, AcatA + ISZ, KCAT, NB, HSZ);
    }

    // --- recurrent steps ---------------------------------------------------
    dim3 gemmBlock(128);
    dim3 gemmGridStep(HSZ / 64, NB / 64);   // (16, 8) = 128 workgroups/step

    for (int t = 0; t < LSTEPS; ++t) {
        __bf16* cur = (t & 1) ? AcatB : AcatA;
        __bf16* nxt = (t & 1) ? AcatA : AcatB;

        // x_t -> bf16 head of cur
        int total = NB * ISZ;
        cvt_f32_bf16_kernel<<<(total + 255) / 256, 256, 0, stream>>>(
            x + (size_t)t * NB * ISZ, ISZ, cur, KCAT, NB, ISZ);

        // h_new = tanh([x_t | h] @ Wcat^T + bsum); bf16 h -> tail of nxt
        float* hf = (t == LSTEPS - 1) ? h_out : nullptr;
        gemm_act_kernel<<<gemmGridStep, gemmBlock, 0, stream>>>(
            cur, KCAT, Wcat, KCAT, bsum,
            /*K=*/KCAT, /*act=*/0,
            /*out_bf=*/nxt + ISZ, /*ldob=*/KCAT,
            /*out_f=*/hf, /*ldof=*/HSZ);
    }

    // --- final fc: y = relu(h @ W_fc^T + b_fc) -----------------------------
    // after the loop (t=255 odd -> nxt == AcatA) final bf16 h is AcatA tail
    dim3 gemmGridFc(ISZ / 64, NB / 64);     // (8, 8)
    gemm_act_kernel<<<gemmGridFc, gemmBlock, 0, stream>>>(
        AcatA + ISZ, KCAT, Wfc_b, HSZ, b_fc,
        /*K=*/HSZ, /*act=*/1,
        /*out_bf=*/nullptr, /*ldob=*/0,
        /*out_f=*/y_out, /*ldof=*/ISZ);
}
